// GRU_78477642432869
// MI455X (gfx1250) — compile-verified, matrix operand order checked by reading, not run on hardware
//
#include <hip/hip_runtime.h>
#include <hip/hip_bf16.h>

#define B_ 256
#define T_ 512
#define I_ 128
#define H_ 256
#define NBLK_SCAN 16
#define LDSP 264   // LDS weight-slab pitch in bf16 elements (528B: 16B aligned, 4-dword bank stagger)

typedef __attribute__((ext_vector_type(16))) __bf16 v16bf;
typedef __attribute__((ext_vector_type(8)))  __bf16 v8bf;
typedef __attribute__((ext_vector_type(8)))  float  v8f;

static __device__ __forceinline__ float sigmoidf_(float x) {
    return 1.0f / (1.0f + __expf(-x));
}
static __device__ __forceinline__ v8f wmma_bf16(v16bf a, v16bf b, v8f c) {
    return __builtin_amdgcn_wmma_f32_16x16x32_bf16(false, a, false, b, (short)0, c, false, false);
}
// 32B-aligned global fragment load
static __device__ __forceinline__ v16bf ld_g16(const __bf16* p) {
    return *(const v16bf*)p;
}
// LDS fragment load (rows only 16B aligned due to pitch stagger) -> two b128 loads
static __device__ __forceinline__ v16bf ld_lds16(const __bf16* p) {
    union { v16bf v; v8bf h[2]; } u;
    u.h[0] = *(const v8bf*)p;
    u.h[1] = *(const v8bf*)(p + 8);
    return u.v;
}

// ---------------------------------------------------------------------------
__global__ void cvt_f32_to_bf16(const float* __restrict__ src,
                                __bf16* __restrict__ dst, int n) {
    int i = blockIdx.x * blockDim.x + threadIdx.x;
    if (i < n) dst[i] = (__bf16)src[i];
}

// Transpose x [B][T][I] f32 -> [T][B][I] bf16 (so projection A-rows are
// 16 consecutive batches at fixed t with 256B row stride).
__global__ void cvt_x_transpose(const float* __restrict__ src,
                                __bf16* __restrict__ dst, int n) {
    int i = blockIdx.x * blockDim.x + threadIdx.x;
    if (i >= n) return;
    int c  = i & (I_ - 1);
    int bt = i >> 7;              // b*T + t
    int t  = bt & (T_ - 1);
    int b  = bt >> 9;
    dst[((size_t)t * B_ + b) * I_ + c] = (__bf16)src[i];
}

// ---------------------------------------------------------------------------
// Input projections in C-fragment layout:
//   proj_g[t][btile][htile][lane*8+v] = bf16( x[b][t][:] . Wg[h][:] + bias_g[h] )
// Wave = (t, btile); loops 3 gates x 16 htiles; K=128 -> 4 WMMAs per tile.
__global__ void __launch_bounds__(256) gru_proj_wmma(
        const __bf16* __restrict__ xbf,          // [T][B][I]
        const __bf16* __restrict__ wr,
        const __bf16* __restrict__ wz,
        const __bf16* __restrict__ wn,
        const float* __restrict__ br, const float* __restrict__ bz,
        const float* __restrict__ bn,
        __bf16* __restrict__ pr, __bf16* __restrict__ pz,
        __bf16* __restrict__ pn) {
    const int lane  = threadIdx.x & 31;
    const int w     = blockIdx.x * (blockDim.x >> 5) + (threadIdx.x >> 5);
    const int btile = w & 15;
    const int t     = w >> 4;
    const int bbase = btile << 4;
    const int l15   = lane & 15;
    const int koff  = (lane >> 4) << 4;

    v16bf a[4];
    const __bf16* xrow = xbf + ((size_t)t * B_ + bbase + l15) * I_ + koff;
#pragma unroll
    for (int k = 0; k < 4; ++k) a[k] = ld_g16(xrow + k * 32);

#pragma unroll
    for (int g = 0; g < 3; ++g) {
        const __bf16* W   = (g == 0) ? wr : (g == 1) ? wz : wn;
        const float*  bia = (g == 0) ? br : (g == 1) ? bz : bn;
        __bf16*       out = (g == 0) ? pr : (g == 1) ? pz : pn;
        for (int nt = 0; nt < 16; ++nt) {
            const int hb = nt << 4;
            v8f acc = {};
            const __bf16* wrow = W + (size_t)(hb + l15) * I_ + koff;
#pragma unroll
            for (int k = 0; k < 4; ++k)
                acc = wmma_bf16(a[k], ld_g16(wrow + k * 32), acc);
            const float bv = bia[hb + l15];          // bias per column N=l15
            v8bf pk;
#pragma unroll
            for (int v = 0; v < 8; ++v) pk[v] = (__bf16)(acc[v] + bv);
            // one 16B store per tile, fully coalesced across the wave
            *(v8bf*)(out + (((size_t)t * 16 + btile) * 16 + nt) * 256 + lane * 8) = pk;
        }
    }
}

// ---------------------------------------------------------------------------
static __device__ __forceinline__ void grid_barrier(unsigned* cnt, unsigned* gen) {
    __threadfence();
    __syncthreads();
    if (threadIdx.x == 0) {
        unsigned g = atomicAdd(gen, 0u);
        if (atomicAdd(cnt, 1u) == NBLK_SCAN - 1) {
            atomicExch(cnt, 0u);
            __threadfence();
            atomicAdd(gen, 1u);
        } else {
            while (atomicAdd(gen, 0u) == g) { __builtin_amdgcn_s_sleep(2); }
        }
    }
    __syncthreads();
}

// Persistent recurrent scan: 16 blocks x 512 threads (16 waves).
// Block = one 16-wide column slab of H (weights LDS-resident all 512 steps);
// wave  = one 16x16 tile of h. Two grid barriers per timestep.
// h state (fp32) kept in C-fragment layout [btile][htile][lane*8+v];
// hbf/rhbf (bf16 A-operands) kept row-major [256][256].
__global__ void __launch_bounds__(512, 1) gru_scan_wmma(
        const __bf16* __restrict__ pr,
        const __bf16* __restrict__ pz,
        const __bf16* __restrict__ pn,
        const __bf16* __restrict__ whr,
        const __bf16* __restrict__ whz,
        const __bf16* __restrict__ wh,
        float* __restrict__ h,          // fragment layout
        __bf16* __restrict__ hbf,       // row-major
        __bf16* __restrict__ rhbf,      // row-major
        unsigned* __restrict__ bar) {
    __shared__ __bf16 slab[3 * 16 * LDSP];   // ~25 KB
    const int tid = threadIdx.x;
    const int hj  = blockIdx.x;
    const int hb  = hj << 4;

    // stage weight slabs: slab[g][n][k] = Wg[hb+n][k]
#pragma unroll
    for (int g = 0; g < 3; ++g) {
        const __bf16* W = (g == 0) ? whr : (g == 1) ? whz : wh;
        for (int idx = tid; idx < 16 * 256; idx += 512) {
            int n = idx >> 8, k = idx & 255;
            slab[g * 16 * LDSP + n * LDSP + k] = W[(size_t)(hb + n) * H_ + k];
        }
    }
    __syncthreads();

    const int lane = tid & 31;
    const int wid  = tid >> 5;             // btile
    const int bib  = wid << 4;
    const int l15  = lane & 15;
    const int koff = (lane >> 4) << 4;
    const int Mo   = (lane >> 4) << 3;

    unsigned* cnt = bar;
    unsigned* gen = bar + 1;

    const __bf16* s_r = &slab[0 * 16 * LDSP + l15 * LDSP + koff];
    const __bf16* s_z = &slab[1 * 16 * LDSP + l15 * LDSP + koff];
    const __bf16* s_n = &slab[2 * 16 * LDSP + l15 * LDSP + koff];

    const __bf16* arow = hbf  + (size_t)(bib + l15) * H_ + koff;
    const __bf16* rrow = rhbf + (size_t)(bib + l15) * H_ + koff;
    const size_t  rbase = (size_t)(bib + Mo) * H_ + hb + l15;    // row-major scatter base
    const size_t  fbase = ((size_t)wid * 16 + hj) * 256 + lane * 8; // fragment base

    for (int t = 0; t < T_; ++t) {
        // ---- phase A: r = sigma(xr + h Whr^T), z = sigma(xz + h Whz^T) ----
        v8f accr = {}, accz = {};
#pragma unroll
        for (int k = 0; k < 8; ++k) {
            v16bf a = ld_g16(arow + k * 32);
            accr = wmma_bf16(a, ld_lds16(s_r + k * 32), accr);
            accz = wmma_bf16(a, ld_lds16(s_z + k * 32), accz);
        }
        const size_t ptile = (((size_t)t * 16 + wid) * 16 + hj) * 256 + lane * 8;
        v8f xrv = __builtin_convertvector(*(const v8bf*)(pr + ptile), v8f);
        v8f xzv = __builtin_convertvector(*(const v8bf*)(pz + ptile), v8f);
        v8f hov = *(const v8f*)(h + fbase);
        v8f zv;
#pragma unroll
        for (int v = 0; v < 8; ++v) {
            float r = sigmoidf_(accr[v] + xrv[v]);
            zv[v]   = sigmoidf_(accz[v] + xzv[v]);
            rhbf[rbase + (size_t)v * H_] = (__bf16)(r * hov[v]);
        }
        grid_barrier(cnt, gen);

        // ---- phase B: ht = tanh(xn + (r.h) Wh^T); h = z.h + (1-z).ht ----
        v8f accn = {};
#pragma unroll
        for (int k = 0; k < 8; ++k) {
            v16bf a = ld_g16(rrow + k * 32);
            accn = wmma_bf16(a, ld_lds16(s_n + k * 32), accn);
        }
        v8f xnv = __builtin_convertvector(*(const v8bf*)(pn + ptile), v8f);
        v8f hnv;
#pragma unroll
        for (int v = 0; v < 8; ++v) {
            float ht = tanhf(accn[v] + xnv[v]);
            float hn = zv[v] * hov[v] + (1.0f - zv[v]) * ht;
            hnv[v] = hn;
            hbf[rbase + (size_t)v * H_] = (__bf16)hn;
        }
        *(v8f*)(h + fbase) = hnv;
        grid_barrier(cnt, gen);
    }
}

// ---------------------------------------------------------------------------
// Head: out = softmax(sigmoid(sigmoid(h) @ w1^T + b1) @ w2^T + b2)
// h arrives in fragment layout [btile][htile][lane*8+v].
__global__ void __launch_bounds__(64) gru_head(
        const float* __restrict__ h,
        const float* __restrict__ w1, const float* __restrict__ b1,
        const float* __restrict__ w2, const float* __restrict__ b2,
        float* __restrict__ out) {
    __shared__ float sh[H_];
    __shared__ float s1[64];
    __shared__ float lg[10];
    const int row = blockIdx.x;
    const int tid = threadIdx.x;
    const int btile = row >> 4;
    const int M     = row & 15;
    const int lhalf = (M >> 3) << 4;   // lane offset for M>=8
    const int v     = M & 7;
    for (int i = tid; i < H_; i += 64) {
        int htile = i >> 4, N = i & 15;
        size_t idx = (((size_t)btile * 16 + htile) * 256) + (size_t)(N + lhalf) * 8 + v;
        sh[i] = sigmoidf_(h[idx]);
    }
    __syncthreads();
    {
        float acc = b1[tid];
        const float* wrow = w1 + (size_t)tid * H_;
        for (int k = 0; k < H_; ++k) acc += sh[k] * wrow[k];
        s1[tid] = sigmoidf_(acc);
    }
    __syncthreads();
    if (tid < 10) {
        float acc = b2[tid];
        const float* wrow = w2 + tid * 64;
        for (int k = 0; k < 64; ++k) acc += s1[k] * wrow[k];
        lg[tid] = acc;
    }
    __syncthreads();
    if (tid == 0) {
        float m = lg[0];
        for (int c = 1; c < 10; ++c) m = fmaxf(m, lg[c]);
        float e[10], s = 0.f;
        for (int c = 0; c < 10; ++c) { e[c] = __expf(lg[c] - m); s += e[c]; }
        float inv = 1.0f / s;
        for (int c = 0; c < 10; ++c) out[(size_t)row * 10 + c] = e[c] * inv;
    }
}

// ---------------------------------------------------------------------------
extern "C" void kernel_launch(void* const* d_in, const int* in_sizes, int n_in,
                              void* d_out, int out_size, void* d_ws, size_t ws_size,
                              hipStream_t stream) {
    (void)in_sizes; (void)n_in; (void)out_size; (void)ws_size;
    const float* x   = (const float*)d_in[0];
    const float* wxr = (const float*)d_in[1];
    const float* whr = (const float*)d_in[2];
    const float* br  = (const float*)d_in[3];
    const float* wxz = (const float*)d_in[4];
    const float* whz = (const float*)d_in[5];
    const float* bz  = (const float*)d_in[6];
    const float* wx  = (const float*)d_in[7];
    const float* wh  = (const float*)d_in[8];
    const float* bn  = (const float*)d_in[9];
    const float* w1  = (const float*)d_in[10];
    const float* b1  = (const float*)d_in[11];
    const float* w2  = (const float*)d_in[12];
    const float* b2  = (const float*)d_in[13];

    char* ws = (char*)d_ws;
    size_t off = 0;
    auto alloc = [&](size_t bytes) -> char* {
        char* p = ws + off;
        off = (off + bytes + 255) & ~(size_t)255;
        return p;
    };
    __bf16* xbf   = (__bf16*)alloc((size_t)B_ * T_ * I_ * 2);   // [T][B][I]
    __bf16* wxr_b = (__bf16*)alloc((size_t)H_ * I_ * 2);
    __bf16* wxz_b = (__bf16*)alloc((size_t)H_ * I_ * 2);
    __bf16* wx_b  = (__bf16*)alloc((size_t)H_ * I_ * 2);
    __bf16* whr_b = (__bf16*)alloc((size_t)H_ * H_ * 2);
    __bf16* whz_b = (__bf16*)alloc((size_t)H_ * H_ * 2);
    __bf16* wh_b  = (__bf16*)alloc((size_t)H_ * H_ * 2);
    __bf16* pr    = (__bf16*)alloc((size_t)T_ * B_ * H_ * 2);   // fragment layout
    __bf16* pz    = (__bf16*)alloc((size_t)T_ * B_ * H_ * 2);
    __bf16* pn    = (__bf16*)alloc((size_t)T_ * B_ * H_ * 2);
    float*  h     = (float*)alloc((size_t)B_ * H_ * 4);         // fragment layout
    __bf16* hbf   = (__bf16*)alloc((size_t)B_ * H_ * 2);        // row-major
    __bf16* rhbf  = (__bf16*)alloc((size_t)B_ * H_ * 2);        // row-major
    unsigned* bar = (unsigned*)alloc(256);

    hipMemsetAsync(h,   0, (size_t)B_ * H_ * 4, stream);   // h0 = 0
    hipMemsetAsync(hbf, 0, (size_t)B_ * H_ * 2, stream);
    hipMemsetAsync(bar, 0, 256, stream);

    const int nx = B_ * T_ * I_;
    cvt_x_transpose<<<(nx + 255) / 256, 256, 0, stream>>>(x, xbf, nx);
    cvt_f32_to_bf16<<<(H_ * I_ + 255) / 256, 256, 0, stream>>>(wxr, wxr_b, H_ * I_);
    cvt_f32_to_bf16<<<(H_ * I_ + 255) / 256, 256, 0, stream>>>(wxz, wxz_b, H_ * I_);
    cvt_f32_to_bf16<<<(H_ * I_ + 255) / 256, 256, 0, stream>>>(wx,  wx_b,  H_ * I_);
    cvt_f32_to_bf16<<<(H_ * H_ + 255) / 256, 256, 0, stream>>>(whr, whr_b, H_ * H_);
    cvt_f32_to_bf16<<<(H_ * H_ + 255) / 256, 256, 0, stream>>>(whz, whz_b, H_ * H_);
    cvt_f32_to_bf16<<<(H_ * H_ + 255) / 256, 256, 0, stream>>>(wh,  wh_b,  H_ * H_);

    // 8192 (t, btile) tiles, 8 waves/block -> 1024 blocks
    gru_proj_wmma<<<(T_ * 16) / 8, 256, 0, stream>>>(
        xbf, wxr_b, wxz_b, wx_b, br, bz, bn, pr, pz, pn);

    gru_scan_wmma<<<NBLK_SCAN, 512, 0, stream>>>(
        pr, pz, pn, whr_b, whz_b, wh_b, h, hbf, rhbf, bar);

    gru_head<<<B_, 64, 0, stream>>>(h, w1, b1, w2, b2, (float*)d_out);
}
